// PhaseHead_30717606101720
// MI455X (gfx1250) — matrix operand order repaired and checked by reading, not compiled, not run on hardware
//
#include <hip/hip_runtime.h>
#include <math.h>

// ---------------------------------------------------------------------------
// Types for CDNA5 WMMA (gfx1250, wave32)
// ---------------------------------------------------------------------------
typedef __bf16 bf16_t;
typedef __attribute__((ext_vector_type(16))) __bf16 v16bf;
typedef __attribute__((ext_vector_type(8)))  __bf16 v8bf;
typedef __attribute__((ext_vector_type(8)))  float  v8f;

static __device__ __forceinline__ bf16_t f2bf(float f) { return (bf16_t)f; }
static __device__ __forceinline__ float  bf2f(bf16_t h) { return (float)h; }

// ---------------------------------------------------------------------------
// Problem constants
// ---------------------------------------------------------------------------
#define NSEQ   64          // bt*st
#define C0     512
#define L0     1024
#define C1     256
#define L1     1024
#define C2     32
#define L2     2048
#define C3     3
#define L3     4096
#define BN_EPS 1e-5f

// K ordering for GEMM: r = tap*C_in + ci  (so a K=32 chunk stays in one tap)
#define K1TOT  1536        // 3*512
#define KS1    48          // 1536/32
#define K2TOT  768         // 3*256
#define KS2    24          // 768/32

// LDS B-tile row stride in halves (16B aligned, bank-skewed)
#define BSTRIDE 40

// ---------------------------------------------------------------------------
// Workspace layout (bytes)
// ---------------------------------------------------------------------------
#define OFF_W1P   ((size_t)0)                 // 16*48*32*16 bf16 = 786432 B
#define OFF_W2P   ((size_t)786432)            // 2*24*32*16 bf16  = 49152 B
#define OFF_BN1   ((size_t)835584)            // 256*2 f32 (scale, shift)
#define OFF_BN2   ((size_t)837632)            // 32*2 f32
#define OFF_FLAGS ((size_t)837888)            // 64 u32
#define OFF_PART  ((size_t)838144)            // 1024 f32
#define OFF_Y1    ((size_t)1048576)           // 64*256*1024 bf16 = 33554432 B
#define OFF_Y2    ((size_t)34603008)          // 64*32*2048 f32  = 16777216 B

// ---------------------------------------------------------------------------
// Fragment K-index for the documented 16-bit A 16x32 layout:
// lanes 0-15 hold K {0..7,16..23}; lanes 16-31 hold K {8..15,24..31}
// ---------------------------------------------------------------------------
static __device__ __forceinline__ int frag_k(int lane, int h) {
  int base = (lane >= 16) ? 8 : 0;
  return (h < 8) ? (base + h) : (16 + base + (h - 8));
}

// ===========================================================================
// K0a: pack w1 [256,512,3] f32 -> A-fragment-ordered bf16
// layout: w1p[((mtile*48+ks)*32+lane)*16+h]
// ===========================================================================
__global__ __launch_bounds__(256) void k_pack_w1(const float* __restrict__ w1,
                                                 bf16_t* __restrict__ w1p) {
  int idx = blockIdx.x * 256 + threadIdx.x;         // < 393216
  int h    = idx & 15;
  int rest = idx >> 4;
  int lane = rest & 31;
  int rest2 = rest >> 5;
  int ks   = rest2 % KS1;
  int mt   = rest2 / KS1;
  int co   = mt * 16 + (lane & 15);
  int r    = ks * 32 + frag_k(lane, h);
  int tap  = r >> 9;                                // /512
  int ci   = r & 511;
  w1p[idx] = f2bf(w1[(co * C0 + ci) * 3 + tap]);
}

// ===========================================================================
// K0b: pack w2 [32,256,3] -> w2p[((mtile*24+ks)*32+lane)*16+h]
// ===========================================================================
__global__ __launch_bounds__(256) void k_pack_w2(const float* __restrict__ w2,
                                                 bf16_t* __restrict__ w2p) {
  int idx = blockIdx.x * 256 + threadIdx.x;         // < 24576
  int h    = idx & 15;
  int rest = idx >> 4;
  int lane = rest & 31;
  int rest2 = rest >> 5;
  int ks   = rest2 % KS2;
  int mt   = rest2 / KS2;
  int co   = mt * 16 + (lane & 15);
  int r    = ks * 32 + frag_k(lane, h);
  int tap  = r >> 8;                                // /256
  int ci   = r & 255;
  w2p[idx] = f2bf(w2[(co * C1 + ci) * 3 + tap]);
}

// ===========================================================================
// K0c: fold BN (eval) into scale/shift pairs
// ===========================================================================
__global__ __launch_bounds__(256) void k_bn(const float* g1, const float* b1,
                                            const float* m1, const float* v1,
                                            const float* g2, const float* b2,
                                            const float* m2, const float* v2,
                                            float* bn1, float* bn2) {
  int i = threadIdx.x;
  if (i < C1) {
    float s = g1[i] * rsqrtf(v1[i] + BN_EPS);
    bn1[2 * i] = s;
    bn1[2 * i + 1] = b1[i] - m1[i] * s;
  }
  if (i < C2) {
    float s = g2[i] * rsqrtf(v2[i] + BN_EPS);
    bn2[2 * i] = s;
    bn2[2 * i + 1] = b2[i] - m2[i] * s;
  }
}

// ===========================================================================
// K1: stage-1 conv as bf16 WMMA GEMM, fused BN+ReLU, y1 stored as bf16
// grid: 1024 WGs (n, 64-col tile); block 256 = 8 waves
// each wave: 2 M-tiles x 4 N-tiles, 48 k-steps of K=32
// ===========================================================================
__global__ __launch_bounds__(256) void k_conv1(const float* __restrict__ x,
                                               const bf16_t* __restrict__ w1p,
                                               const float* __restrict__ bn1,
                                               bf16_t* __restrict__ y1) {
  __shared__ bf16_t ldsB[64 * BSTRIDE];             // 5120 B

  const int n    = blockIdx.x >> 4;
  const int t0   = (blockIdx.x & 15) * 64;
  const int tid  = threadIdx.x;
  const int wave = tid >> 5;
  const int lane = tid & 31;
  const int m0   = wave * 2;                        // this wave's first M-tile

  v8f acc[2][4];
#pragma unroll
  for (int m = 0; m < 2; ++m)
#pragma unroll
    for (int nt = 0; nt < 4; ++nt) acc[m][nt] = (v8f)0.0f;

  for (int ks = 0; ks < KS1; ++ks) {
    const int tap = ks >> 4;
    const int ci0 = (ks & 15) * 32;

    // ---- stage im2col B tile into LDS (f32 -> bf16 in registers) ----
    {
      const int kk = tid >> 3;                      // 0..31
      const int j0 = (tid & 7) * 8;                 // 0..56
      const float* xrow = x + ((size_t)n * C0 + (ci0 + kk)) * L0;
      const int pb = t0 + tap * 8 - 8 + j0;         // shifted, may reflect
      float v[8];
      if (pb >= 0 && pb + 7 < L0) {
        const float4 a = *(const float4*)(xrow + pb);
        const float4 b = *(const float4*)(xrow + pb + 4);
        v[0]=a.x; v[1]=a.y; v[2]=a.z; v[3]=a.w;
        v[4]=b.x; v[5]=b.y; v[6]=b.z; v[7]=b.w;
      } else {
#pragma unroll
        for (int i = 0; i < 8; ++i) {
          int p = pb + i;
          p = (p < 0) ? -p : ((p > L0 - 1) ? (2 * (L0 - 1) - p) : p);
          v[i] = xrow[p];
        }
      }
#pragma unroll
      for (int i = 0; i < 8; ++i)
        ldsB[(j0 + i) * BSTRIDE + kk] = f2bf(v[i]);
    }
    __syncthreads();

    // ---- A fragments (pre-packed, per-lane aligned vector loads) ----
    v16bf afr[2];
#pragma unroll
    for (int m = 0; m < 2; ++m)
      afr[m] = *(const v16bf*)(w1p + (((size_t)(m0 + m) * KS1 + ks) * 32 + lane) * 16);

    // ---- B fragments from LDS + 8 WMMAs ----
    const int kb = (lane < 16) ? 0 : 8;
#pragma unroll
    for (int nt = 0; nt < 4; ++nt) {
      const bf16_t* bp = ldsB + (nt * 16 + (lane & 15)) * BSTRIDE + kb;
      v8bf lo = *(const v8bf*)bp;
      v8bf hi = *(const v8bf*)(bp + 16);
      v16bf bfr;
#pragma unroll
      for (int i = 0; i < 8; ++i) { bfr[i] = lo[i]; bfr[i + 8] = hi[i]; }
      acc[0][nt] = __builtin_amdgcn_wmma_f32_16x16x32_bf16(
          false, afr[0], false, bfr, (short)0, acc[0][nt], false, false);
      acc[1][nt] = __builtin_amdgcn_wmma_f32_16x16x32_bf16(
          false, afr[1], false, bfr, (short)0, acc[1][nt], false, false);
    }
    __syncthreads();
  }

  // ---- epilogue: BN + ReLU, store bf16 (documented C/D layout) ----
  const int rowoff = (lane >> 4) * 8;
  const int col    = lane & 15;
#pragma unroll
  for (int m = 0; m < 2; ++m) {
#pragma unroll
    for (int v = 0; v < 8; ++v) {
      const int co = (m0 + m) * 16 + rowoff + v;
      const float s = bn1[2 * co], o = bn1[2 * co + 1];
      bf16_t* yrow = y1 + ((size_t)n * C1 + co) * L1 + t0;
#pragma unroll
      for (int nt = 0; nt < 4; ++nt) {
        float val = fmaxf(acc[m][nt][v] * s + o, 0.0f);
        yrow[nt * 16 + col] = f2bf(val);
      }
    }
  }
}

// ===========================================================================
// K2: upsample(y1) fused into im2col, stage-2 WMMA GEMM, BN+ReLU, y2 f32
// grid: 1024 WGs (n, 128-col tile); block 256 = 8 waves
// each wave: 2 M-tiles x 1 N-tile (its own 16 cols), 24 k-steps
// ===========================================================================
__global__ __launch_bounds__(256) void k_conv2(const bf16_t* __restrict__ y1,
                                               const bf16_t* __restrict__ w2p,
                                               const float* __restrict__ bn2,
                                               float* __restrict__ y2) {
  __shared__ bf16_t ldsB[128 * BSTRIDE];            // 10240 B

  const int n    = blockIdx.x >> 4;
  const int t0   = (blockIdx.x & 15) * 128;
  const int tid  = threadIdx.x;
  const int wave = tid >> 5;
  const int lane = tid & 31;

  v8f acc[2];
  acc[0] = (v8f)0.0f; acc[1] = (v8f)0.0f;

  for (int ks = 0; ks < KS2; ++ks) {
    const int tap = ks >> 3;
    const int ci0 = (ks & 7) * 32;

    // ---- stage upsampled im2col tile into LDS ----
    {
      const int kk = tid >> 3;                      // 0..31
      const int j0 = (tid & 7) * 16;                // 0..112
      const bf16_t* yrow = y1 + ((size_t)n * C1 + (ci0 + kk)) * L1;
#pragma unroll 4
      for (int i = 0; i < 16; ++i) {
        int p = t0 + j0 + i + tap * 16 - 16;        // position in u1 (len 2048)
        p = (p < 0) ? -p : ((p > L2 - 1) ? (2 * (L2 - 1) - p) : p);
        const int h = p >> 1;
        float u;
        if ((p & 1) == 0) {
          u = (h == 0) ? bf2f(yrow[0])
                       : 0.25f * bf2f(yrow[h - 1]) + 0.75f * bf2f(yrow[h]);
        } else {
          const int h1 = (h + 1 > L1 - 1) ? (L1 - 1) : (h + 1);
          u = 0.75f * bf2f(yrow[h]) + 0.25f * bf2f(yrow[h1]);
        }
        ldsB[(j0 + i) * BSTRIDE + kk] = f2bf(u);
      }
    }
    __syncthreads();

    v16bf afr[2];
#pragma unroll
    for (int m = 0; m < 2; ++m)
      afr[m] = *(const v16bf*)(w2p + (((size_t)m * KS2 + ks) * 32 + lane) * 16);

    const int kb = (lane < 16) ? 0 : 8;
    const bf16_t* bp = ldsB + (wave * 16 + (lane & 15)) * BSTRIDE + kb;
    v8bf lo = *(const v8bf*)bp;
    v8bf hi = *(const v8bf*)(bp + 16);
    v16bf bfr;
#pragma unroll
    for (int i = 0; i < 8; ++i) { bfr[i] = lo[i]; bfr[i + 8] = hi[i]; }
    acc[0] = __builtin_amdgcn_wmma_f32_16x16x32_bf16(
        false, afr[0], false, bfr, (short)0, acc[0], false, false);
    acc[1] = __builtin_amdgcn_wmma_f32_16x16x32_bf16(
        false, afr[1], false, bfr, (short)0, acc[1], false, false);
    __syncthreads();
  }

  const int rowoff = (lane >> 4) * 8;
  const int col    = lane & 15;
  const int t      = t0 + wave * 16 + col;
#pragma unroll
  for (int m = 0; m < 2; ++m) {
#pragma unroll
    for (int v = 0; v < 8; ++v) {
      const int co = m * 16 + rowoff + v;
      const float s = bn2[2 * co], o = bn2[2 * co + 1];
      y2[((size_t)n * C2 + co) * L2 + t] = fmaxf(acc[m][v] * s + o, 0.0f);
    }
  }
}

// ===========================================================================
// K_valid: valid[b,s] = any(targets[b,:,:,s] != 0); 64 blocks
// ===========================================================================
__global__ __launch_bounds__(256) void k_valid(const float* __restrict__ tg,
                                               unsigned* __restrict__ flags) {
  __shared__ int f;
  const int b = blockIdx.x >> 2, s = blockIdx.x & 3;
  if (threadIdx.x == 0) f = 0;
  __syncthreads();
  int local = 0;
  for (int idx = threadIdx.x; idx < C3 * L3; idx += 256) {
    const int c = idx / L3, t = idx - c * L3;
    if (tg[(((size_t)b * C3 + c) * L3 + t) * 4 + s] != 0.0f) local = 1;
  }
  if (local) atomicOr(&f, 1);
  __syncthreads();
  if (threadIdx.x == 0) flags[blockIdx.x] = (unsigned)f;
}

// ===========================================================================
// K3: upsample(y2) fused into stage-3 conv (32->3), transpose-store output,
//     per-block partial of masked CE numerator (deterministic reduction)
// grid: 1024 blocks x 256 threads, one (n,t) per thread
// ===========================================================================
__global__ __launch_bounds__(256) void k_conv3(const float* __restrict__ y2,
                                               const float* __restrict__ w_out,
                                               const float* __restrict__ b_out,
                                               const float* __restrict__ tg,
                                               float* __restrict__ outx,
                                               float* __restrict__ partials) {
  __shared__ float wsm[C3 * C2 * 3];
  __shared__ float bsm[C3];
  __shared__ float red[256];
  const int tid = threadIdx.x;
  for (int i = tid; i < C3 * C2 * 3; i += 256) wsm[i] = w_out[i];
  if (tid < C3) bsm[tid] = b_out[tid];
  __syncthreads();

  const int gid = blockIdx.x * 256 + tid;
  const int n = gid >> 12;                          // /4096
  const int t = gid & (L3 - 1);
  const int b = n >> 2, s = n & 3;
  const float* ybase = y2 + (size_t)n * C2 * L2;

  float a0 = bsm[0], a1 = bsm[1], a2 = bsm[2];
#pragma unroll
  for (int k = 0; k < 3; ++k) {
    int p = t + 32 * k - 32;                        // position in u2 (len 4096)
    p = (p < 0) ? -p : ((p > L3 - 1) ? (2 * (L3 - 1) - p) : p);
    const int h = p >> 1;
    const int odd = p & 1;
    const int h1 = (h + 1 > L2 - 1) ? (L2 - 1) : (h + 1);
    for (int ci = 0; ci < C2; ++ci) {
      const float* yr = ybase + (size_t)ci * L2;
      float u;
      if (!odd) u = (h == 0) ? yr[0] : 0.25f * yr[h - 1] + 0.75f * yr[h];
      else      u = 0.75f * yr[h] + 0.25f * yr[h1];
      a0 += wsm[(0 * C2 + ci) * 3 + k] * u;
      a1 += wsm[(1 * C2 + ci) * 3 + k] * u;
      a2 += wsm[(2 * C2 + ci) * 3 + k] * u;
    }
  }

  // transpose-store: out[b, c, t, s]
  const size_t ob = (((size_t)b * C3) * L3 + t) * 4 + s;
  outx[ob]              = a0;
  outx[ob + (size_t)L3 * 4]     = a1;
  outx[ob + (size_t)2 * L3 * 4] = a2;

  // CE numerator: sum_c targets * (lse - x_c)
  const float mx = fmaxf(a0, fmaxf(a1, a2));
  const float lse = mx + logf(expf(a0 - mx) + expf(a1 - mx) + expf(a2 - mx));
  const size_t tb = (((size_t)b * C3) * L3 + t) * 4 + s;
  float part = tg[tb] * (lse - a0)
             + tg[tb + (size_t)L3 * 4] * (lse - a1)
             + tg[tb + (size_t)2 * L3 * 4] * (lse - a2);

  red[tid] = part;
  __syncthreads();
  for (int off = 128; off > 0; off >>= 1) {
    if (tid < off) red[tid] += red[tid + off];
    __syncthreads();
  }
  if (tid == 0) partials[blockIdx.x] = red[0];
}

// ===========================================================================
// K4: final deterministic reduction -> loss scalar
// ===========================================================================
__global__ __launch_bounds__(256) void k_final(const float* __restrict__ partials,
                                               const unsigned* __restrict__ flags,
                                               float* __restrict__ loss_out) {
  __shared__ float red[256];
  const int tid = threadIdx.x;
  float local = partials[tid] + partials[tid + 256] +
                partials[tid + 512] + partials[tid + 768];
  red[tid] = local;
  __syncthreads();
  for (int off = 128; off > 0; off >>= 1) {
    if (tid < off) red[tid] += red[tid + off];
    __syncthreads();
  }
  if (tid == 0) {
    int num = 0;
    for (int i = 0; i < 64; ++i) num += (int)flags[i];
    loss_out[0] = red[0] / ((float)num * (float)L3);
  }
}

// ===========================================================================
// Host launcher
// ===========================================================================
extern "C" void kernel_launch(void* const* d_in, const int* in_sizes, int n_in,
                              void* d_out, int out_size, void* d_ws, size_t ws_size,
                              hipStream_t stream) {
  const float* fusion  = (const float*)d_in[0];
  const float* targets = (const float*)d_in[1];
  const float* w1      = (const float*)d_in[2];
  const float* bn1_g   = (const float*)d_in[3];
  const float* bn1_b   = (const float*)d_in[4];
  const float* bn1_m   = (const float*)d_in[5];
  const float* bn1_v   = (const float*)d_in[6];
  const float* w2      = (const float*)d_in[7];
  const float* bn2_g   = (const float*)d_in[8];
  const float* bn2_b   = (const float*)d_in[9];
  const float* bn2_m   = (const float*)d_in[10];
  const float* bn2_v   = (const float*)d_in[11];
  const float* w_out   = (const float*)d_in[12];
  const float* b_out   = (const float*)d_in[13];

  char* ws = (char*)d_ws;
  bf16_t*   w1p   = (bf16_t*)(ws + OFF_W1P);
  bf16_t*   w2p   = (bf16_t*)(ws + OFF_W2P);
  float*    bn1   = (float*)(ws + OFF_BN1);
  float*    bn2   = (float*)(ws + OFF_BN2);
  unsigned* flags = (unsigned*)(ws + OFF_FLAGS);
  float*    part  = (float*)(ws + OFF_PART);
  bf16_t*   y1    = (bf16_t*)(ws + OFF_Y1);
  float*    y2    = (float*)(ws + OFF_Y2);

  float* outx = (float*)d_out;
  float* loss = outx + (size_t)16 * C3 * L3 * 4;    // index 786432

  k_pack_w1<<<1536, 256, 0, stream>>>(w1, w1p);
  k_pack_w2<<<96, 256, 0, stream>>>(w2, w2p);
  k_bn<<<1, 256, 0, stream>>>(bn1_g, bn1_b, bn1_m, bn1_v,
                              bn2_g, bn2_b, bn2_m, bn2_v, bn1, bn2);
  k_conv1<<<1024, 256, 0, stream>>>(fusion, w1p, bn1, y1);
  k_conv2<<<1024, 256, 0, stream>>>(y1, w2p, bn2, y2);
  k_valid<<<64, 256, 0, stream>>>(targets, flags);
  k_conv3<<<1024, 256, 0, stream>>>(y2, w_out, b_out, targets, outx, part);
  k_final<<<1, 256, 0, stream>>>(part, flags, loss);
}
